// MHCN_10737418240849
// MI455X (gfx1250) — compile-verified, matrix operand order checked by reading, not run on hardware
//
#include <hip/hip_runtime.h>
#include <hip/hip_bf16.h>

// ---------------------------------------------------------------------------
// MHCN forward for MI455X (gfx1250, wave32).
// D = 128 fixed. One wave (32 lanes) holds one 128-float row as float4/lane.
// Gating GEMMs use v_wmma_f32_16x16x32_bf16 with fragment-ready LDS layout for
// the weight operand (2x ds_load_b128 per fragment) and direct global b128
// loads for the streamed embedding operand. Everything else is HBM-bound
// streaming / atomic-scatter SpMM.
// ---------------------------------------------------------------------------

#define DIMD 128

typedef __attribute__((ext_vector_type(16))) __bf16 v16bf;
typedef __attribute__((ext_vector_type(8)))  __bf16 v8bf;
typedef __attribute__((ext_vector_type(8)))  float  v8f;

__device__ __forceinline__ float wave_sum32(float v) {
  #pragma unroll
  for (int o = 16; o > 0; o >>= 1) v += __shfl_xor(v, o, 32);
  return v;
}

// ---------------------------------------------------------------------------
// Gating: out[g][u][n] = user_emb[u][n] * sigmoid( (user_emb @ W_g)[u][n] + b_g[n] )
// Block: 256 threads = 8 waves. Tile: 64 users x 128 cols, one gate per blockIdx.y.
// Wave w: m-subtile = w%4 (16 users), n-half = w/4 (4 n-tiles of 16 columns).
//
// WMMA 16-bit operand lane layout (ISA 7.12.2): lane-half 0 holds K 0-7 (V0-3)
// and 16-23 (V4-7); lane-half 1 holds K 8-15 and 24-31:
//   klocal(e, half) = e + 8*((e>>3) + half)
// ---------------------------------------------------------------------------
__global__ __launch_bounds__(256) void gating_wmma(
    const float* __restrict__ user_emb,   // [U,128]
    const float* __restrict__ gating_w,   // [4,128,128] row-major (k,n)
    const float* __restrict__ gating_b,   // [4,128]
    float* __restrict__ out,              // [4][U][128]
    int U)
{
  // Fragment-ready B storage: frag (ntile*4 + kstep) -> 32 lanes x 16 bf16.
  __shared__ __align__(16) __bf16 sB[DIMD * DIMD];   // 32 KB

  const int gate = blockIdx.y;
  const int m0   = blockIdx.x * 64;
  const float* __restrict__ W = gating_w + (size_t)gate * DIMD * DIMD;

  // Stage W (f32 -> bf16) directly into WMMA fragment layout.
  // One b32 word per iteration = elements (2j, 2j+1) of one lane's fragment.
  for (int w = threadIdx.x; w < 8192; w += 256) {
    int j      = w & 7;           // element pair within lane fragment
    int lane_  = (w >> 3) & 31;   // consuming lane
    int ntk    = w >> 8;          // ntile*4 + kstep  (0..31)
    int kstep  = ntk & 3;
    int ntile  = ntk >> 2;
    int hb_    = lane_ >> 4;
    int idx_   = lane_ & 15;
    int e      = 2 * j;
    int klocal = e + 8 * ((e >> 3) + hb_);
    int k      = kstep * 32 + klocal;
    int n      = ntile * 16 + idx_;
    union { __bf16 h[2]; unsigned u; } pk;
    pk.h[0] = (__bf16)W[(size_t)k * DIMD + n];
    pk.h[1] = (__bf16)W[(size_t)(k + 1) * DIMD + n];
    ((unsigned*)sB)[w] = pk.u;    // dest element index == 2*w
  }
  __syncthreads();

  const int wave = threadIdx.x >> 5;
  const int lane = threadIdx.x & 31;
  const int hb   = lane >> 4;       // lane half (0/1)
  const int idx  = lane & 15;       // position within half
  const int msub = wave & 3;        // 16-user subtile
  const int ngrp = wave >> 2;       // 0/1 -> n-tiles [0..3] or [4..7]

  // A row for this lane (clamped: out-of-range rows only feed guarded outputs).
  int urow = m0 + 16 * msub + idx;
  if (urow > U - 1) urow = U - 1;
  const float* __restrict__ arow = user_emb + (size_t)urow * DIMD;

  v8f acc0 = {}, acc1 = {}, acc2 = {}, acc3 = {};

  #pragma unroll
  for (int kk = 0; kk < DIMD; kk += 32) {
    // A fragment: two contiguous 8-float runs per lane, four global b128 loads.
    float fv[16];
    *(float4*)(fv + 0)  = *(const float4*)(arow + kk + 8 * hb + 0);
    *(float4*)(fv + 4)  = *(const float4*)(arow + kk + 8 * hb + 4);
    *(float4*)(fv + 8)  = *(const float4*)(arow + kk + 16 + 8 * hb + 0);
    *(float4*)(fv + 12) = *(const float4*)(arow + kk + 16 + 8 * hb + 4);
    v16bf afrag;
    #pragma unroll
    for (int e = 0; e < 16; ++e) afrag[e] = (__bf16)fv[e];

    // 4 B fragments: 2x ds_load_b128 each from fragment-ready LDS.
    #pragma unroll
    for (int t = 0; t < 4; ++t) {
      int ntk = ((ngrp * 4 + t) * 4 + (kk >> 5));
      int fb  = (ntk * 32 + lane) * 2;          // v8bf units
      v8bf blo = ((const v8bf*)sB)[fb + 0];
      v8bf bhi = ((const v8bf*)sB)[fb + 1];
      v16bf bfrag = __builtin_shufflevector(blo, bhi,
          0, 1, 2, 3, 4, 5, 6, 7, 8, 9, 10, 11, 12, 13, 14, 15);
      v8f* accp = (t == 0) ? &acc0 : (t == 1) ? &acc1 : (t == 2) ? &acc2 : &acc3;
      *accp = __builtin_amdgcn_wmma_f32_16x16x32_bf16(
          false, afrag, false, bfrag, (short)0, *accp, false, false);
    }
  }

  // Epilogue: sigmoid gate against the original fp32 embedding.
  const float* __restrict__ bvec = gating_b + gate * DIMD;
  float* __restrict__ o = out + (size_t)gate * (size_t)U * DIMD;
  #pragma unroll
  for (int t = 0; t < 4; ++t) {
    v8f acc = (t == 0) ? acc0 : (t == 1) ? acc1 : (t == 2) ? acc2 : acc3;
    int n = ngrp * 64 + t * 16 + idx;
    float bias = bvec[n];
    #pragma unroll
    for (int r = 0; r < 8; ++r) {
      int m = 16 * msub + r + 8 * hb;   // C/D layout: VGPR r + lane-half -> row
      int u = m0 + m;
      if (u < U) {
        float ue = user_emb[(size_t)u * DIMD + n];
        float g  = acc[r] + bias;
        float sg = 1.0f / (1.0f + __expf(-g));
        o[(size_t)u * DIMD + n] = ue * sg;
      }
    }
  }
}

// ---------------------------------------------------------------------------
// COO SpMM: y[row[e],:] += val[e] * x[col[e],:], one wave per edge, fp32 atomics.
// ---------------------------------------------------------------------------
__global__ __launch_bounds__(256) void spmm_atomic(
    const int* __restrict__ row, const int* __restrict__ col,
    const float* __restrict__ val, const float* __restrict__ x,
    float* __restrict__ y, int E)
{
  int gid  = blockIdx.x * blockDim.x + threadIdx.x;
  int e    = gid >> 5;
  int lane = gid & 31;
  if (e >= E) return;
  // stream-ahead hints for the edge arrays (global_prefetch_b8)
  __builtin_prefetch(row + e + 2048, 0, 1);
  __builtin_prefetch(col + e + 2048, 0, 1);
  __builtin_prefetch(val + e + 2048, 0, 1);
  int   r = row[e];
  int   c = col[e];
  float v = val[e];
  float4 d = ((const float4*)(x + (size_t)c * DIMD))[lane];
  float* yp = y + (size_t)r * DIMD + lane * 4;
  atomicAdd(yp + 0, v * d.x);
  atomicAdd(yp + 1, v * d.y);
  atomicAdd(yp + 2, v * d.z);
  atomicAdd(yp + 3, v * d.w);
}

// ---------------------------------------------------------------------------
// acc[r,:] += x[r,:] / max(||x[r,:]||, 1e-12), one wave per row.
// ---------------------------------------------------------------------------
__global__ __launch_bounds__(256) void l2norm_acc(
    const float* __restrict__ xin, float* __restrict__ acc, int n)
{
  int gid  = blockIdx.x * blockDim.x + threadIdx.x;
  int rw   = gid >> 5;
  int lane = gid & 31;
  if (rw >= n) return;
  float4 d = ((const float4*)(xin + (size_t)rw * DIMD))[lane];
  float s   = wave_sum32(d.x * d.x + d.y * d.y + d.z * d.z + d.w * d.w);
  float inv = 1.0f / fmaxf(sqrtf(s), 1e-12f);
  float4* ap = (float4*)(acc + (size_t)rw * DIMD) + lane;
  float4 a = *ap;
  a.x += d.x * inv; a.y += d.y * inv; a.z += d.z * inv; a.w += d.w * inv;
  *ap = a;
}

// ---------------------------------------------------------------------------
// Channel attention + mix. t_c = e_c[u,:] . wv ; softmax over 3 channels;
// out = sum_c sc_c * e_c + 0.5 * extra. One wave per user.
// ---------------------------------------------------------------------------
__global__ __launch_bounds__(256) void attn_mix(
    const float* __restrict__ e1, const float* __restrict__ e2,
    const float* __restrict__ e3, const float* __restrict__ extra,
    const float* __restrict__ wv, float* __restrict__ out, int n)
{
  int gid  = blockIdx.x * blockDim.x + threadIdx.x;
  int rw   = gid >> 5;
  int lane = gid & 31;
  if (rw >= n) return;
  size_t base = (size_t)rw * DIMD;
  float4 w4 = ((const float4*)wv)[lane];
  float4 d1 = ((const float4*)(e1 + base))[lane];
  float4 d2 = ((const float4*)(e2 + base))[lane];
  float4 d3 = ((const float4*)(e3 + base))[lane];
  float t1 = wave_sum32(d1.x * w4.x + d1.y * w4.y + d1.z * w4.z + d1.w * w4.w);
  float t2 = wave_sum32(d2.x * w4.x + d2.y * w4.y + d2.z * w4.z + d2.w * w4.w);
  float t3 = wave_sum32(d3.x * w4.x + d3.y * w4.y + d3.z * w4.z + d3.w * w4.w);
  float m  = fmaxf(t1, fmaxf(t2, t3));
  float s1 = __expf(t1 - m), s2 = __expf(t2 - m), s3 = __expf(t3 - m);
  float is = 1.0f / (s1 + s2 + s3);
  s1 *= is; s2 *= is; s3 *= is;
  float4 ex = ((const float4*)(extra + base))[lane];
  float4 o;
  o.x = s1 * d1.x + s2 * d2.x + s3 * d3.x + 0.5f * ex.x;
  o.y = s1 * d1.y + s2 * d2.y + s3 * d3.y + 0.5f * ex.y;
  o.z = s1 * d1.z + s2 * d2.z + s3 * d3.z + 0.5f * ex.z;
  o.w = s1 * d1.w + s2 * d2.w + s3 * d3.w + 0.5f * ex.w;
  ((float4*)(out + base))[lane] = o;
}

// wv[d] = sum_n attn_mat[d,n] * attn[n]   (collapses e@attn_mat . attn to e . wv)
__global__ void attn_vec(const float* __restrict__ am,
                         const float* __restrict__ at,
                         float* __restrict__ wv)
{
  int d = threadIdx.x;  // blockDim = 128
  float s = 0.0f;
  #pragma unroll 4
  for (int n = 0; n < DIMD; ++n) s += am[d * DIMD + n] * at[n];
  wv[d] = s;
}

__global__ __launch_bounds__(256) void fillz(float* __restrict__ p, long n)
{
  long i  = ((long)blockIdx.x * blockDim.x + threadIdx.x);
  long i4 = i * 4;
  if (i4 < n) *(float4*)(p + i4) = make_float4(0.f, 0.f, 0.f, 0.f);
}

// ---------------------------------------------------------------------------
extern "C" void kernel_launch(void* const* d_in, const int* in_sizes, int n_in,
                              void* d_out, int out_size, void* d_ws, size_t ws_size,
                              hipStream_t stream) {
  const float* user_emb = (const float*)d_in[0];
  const float* item_emb = (const float*)d_in[1];
  const float* gating_w = (const float*)d_in[2];
  const float* gating_b = (const float*)d_in[3];
  const float* attn     = (const float*)d_in[4];
  const float* attn_mat = (const float*)d_in[5];
  const int*   hrow[3]  = {(const int*)d_in[6],  (const int*)d_in[9],  (const int*)d_in[12]};
  const int*   hcol[3]  = {(const int*)d_in[7],  (const int*)d_in[10], (const int*)d_in[13]};
  const float* hval[3]  = {(const float*)d_in[8],(const float*)d_in[11],(const float*)d_in[14]};
  const int*   r_row    = (const int*)d_in[15];
  const int*   r_col    = (const int*)d_in[16];
  const float* r_val    = (const float*)d_in[17];

  const int U = in_sizes[0] / DIMD;
  const int I = in_sizes[1] / DIMD;
  const int E = in_sizes[6];
  const size_t UD = (size_t)U * DIMD;
  const size_t ID = (size_t)I * DIMD;

  // Workspace layout (floats)
  float* cur = (float*)d_ws;
  float* gates   = cur; cur += 4 * UD;  // c1,c2,c3,simple (current)
  float* gates_b = cur; cur += 4 * UD;  // ping-pong targets
  float* accs    = cur; cur += 4 * UD;  // a1,a2,a3,a_s
  float* mixed   = cur; cur += UD;
  float* itemA   = cur; cur += ID;
  float* itemB   = cur; cur += ID;
  float* wvec    = cur; cur += 256;

  float* out_user = (float*)d_out;
  float* out_item = (float*)d_out + UD;

  const int rowsGridU = (U + 7) / 8;         // one wave per row, 8 rows / block
  const int rowsGridI = (I + 7) / 8;
  const int spmmGrid  = (E + 7) / 8;         // one wave per edge
  const int fillGridU = (int)((UD / 4 + 255) / 256);
  const int fillGridI = (int)((ID / 4 + 255) / 256);

  // 1) Gating GEMMs (WMMA bf16) -> c1,c2,c3,simple
  gating_wmma<<<dim3((U + 63) / 64, 4), 256, 0, stream>>>(
      user_emb, gating_w, gating_b, gates, U);

  // 2) Attention weight vector wv = attn_mat @ attn^T
  attn_vec<<<1, DIMD, 0, stream>>>(attn_mat, attn, wvec);

  // 3) Initialize accumulators: a1..a3=c1..c3, a_s=simple ; a_i=item_emb
  hipMemcpyAsync(accs, gates, 4 * UD * sizeof(float), hipMemcpyDeviceToDevice, stream);
  hipMemcpyAsync(out_item, item_emb, ID * sizeof(float), hipMemcpyDeviceToDevice, stream);

  float* c_cur[4]; float* c_nxt[4];
  for (int ch = 0; ch < 4; ++ch) { c_cur[ch] = gates + ch * UD; c_nxt[ch] = gates_b + ch * UD; }
  const float* item_cur = item_emb;
  float* item_nxt = itemA;

  for (int layer = 0; layer < 2; ++layer) {
    // mixed = attn(c1,c2,c3) + simple/2
    attn_mix<<<rowsGridU, 256, 0, stream>>>(c_cur[0], c_cur[1], c_cur[2], c_cur[3],
                                            wvec, mixed, U);
    // user channels through hypergraphs
    for (int ch = 0; ch < 3; ++ch) {
      fillz<<<fillGridU, 256, 0, stream>>>(c_nxt[ch], (long)UD);
      spmm_atomic<<<spmmGrid, 256, 0, stream>>>(hrow[ch], hcol[ch], hval[ch],
                                                c_cur[ch], c_nxt[ch], E);
      l2norm_acc<<<rowsGridU, 256, 0, stream>>>(c_nxt[ch], accs + ch * UD, U);
    }
    // new_item = R^T @ mixed ; a_i += l2(new_item)
    fillz<<<fillGridI, 256, 0, stream>>>(item_nxt, (long)ID);
    spmm_atomic<<<spmmGrid, 256, 0, stream>>>(r_col, r_row, r_val, mixed, item_nxt, E);
    l2norm_acc<<<rowsGridI, 256, 0, stream>>>(item_nxt, out_item, I);
    // simple = R @ item ; a_s += l2(simple)
    fillz<<<fillGridU, 256, 0, stream>>>(c_nxt[3], (long)UD);
    spmm_atomic<<<spmmGrid, 256, 0, stream>>>(r_row, r_col, r_val, item_cur, c_nxt[3], E);
    l2norm_acc<<<rowsGridU, 256, 0, stream>>>(c_nxt[3], accs + 3 * UD, U);
    // swap ping-pong
    for (int ch = 0; ch < 4; ++ch) { float* t = c_cur[ch]; c_cur[ch] = c_nxt[ch]; c_nxt[ch] = t; }
    item_cur = item_nxt;
    item_nxt = (layer == 0) ? itemB : itemA;
  }

  // 4) Final: user_all = attn(a1,a2,a3) + a_s/2
  attn_mix<<<rowsGridU, 256, 0, stream>>>(accs, accs + UD, accs + 2 * UD, accs + 3 * UD,
                                          wvec, out_user, U);
}
